// DeepAR_57982058496384
// MI455X (gfx1250) — matrix operand order repaired and checked by reading, not compiled, not run on hardware
//
#include <hip/hip_runtime.h>
#include <cstdint>
#include <cstddef>

// ---- shapes ----
#define B_    1024
#define SEQ_  192
#define PRED_ 96
#define W_    288
#define HID_  128
#define G_    512          // 4*HID gates
#define K0_   224          // [x(67) pad->96 | h1(128)]
#define K1_   256          // [h1(128) | h2(128)]

typedef __bf16 bf16_t;
typedef bf16_t v16bf __attribute__((ext_vector_type(16)));
typedef float  v8f   __attribute__((ext_vector_type(8)));

struct alignas(16) U128 { unsigned int a, b, c, d; };
struct alignas(16) F128 { float a, b, c, d; };

union FragAB { v16bf v; U128 q[2]; };
union FragC  { v8f   v; float f[8]; };

__device__ __forceinline__ unsigned short f2bf(float f) {
    unsigned int u = __float_as_uint(f);
    unsigned int r = (u + 0x7FFFu + ((u >> 16) & 1u)) >> 16;   // RNE
    return (unsigned short)r;
}
__device__ __forceinline__ float bf2f(unsigned short u) {
    return __uint_as_float(((unsigned int)u) << 16);
}

// ---- fast activations: CDNA5 has a hardware v_tanh_f32 transcendental ----
#if __has_builtin(__builtin_amdgcn_tanhf)
__device__ __forceinline__ float ftanh(float x) { return __builtin_amdgcn_tanhf(x); }
#else
// branch-free fallback: tanh(x) = 1 - 2/(exp2(x*2*log2e)+1), clamped range
__device__ __forceinline__ float ftanh(float x) {
    float xc = fminf(fmaxf(x, -10.0f), 10.0f);
    float e  = __builtin_amdgcn_exp2f(xc * 2.8853900817779268f);  // exp(2x)
    return (e - 1.0f) / (e + 1.0f);
}
#endif
__device__ __forceinline__ float sigm(float x) {
    return __builtin_fmaf(ftanh(x * 0.5f), 0.5f, 0.5f);
}

// ---------------------------------------------------------------------------
// Kernel 1: per-series instance norm (detached mean, ddof=0 var) + teacher seq
// ---------------------------------------------------------------------------
__global__ void k_norm(const float* __restrict__ x_enc,
                       float* __restrict__ teacher,
                       float* __restrict__ nmean, float* __restrict__ nstd) {
    const int b = blockIdx.x, tid = threadIdx.x;
    __shared__ float red[256];
    __shared__ float s_mean, s_std;
    float v = (tid < SEQ_) ? x_enc[b * SEQ_ + tid] : 0.0f;
    red[tid] = v; __syncthreads();
    for (int s = 128; s > 0; s >>= 1) { if (tid < s) red[tid] += red[tid + s]; __syncthreads(); }
    if (tid == 0) s_mean = red[0] / (float)SEQ_;
    __syncthreads();
    float m = s_mean;
    float d = (tid < SEQ_) ? (v - m) : 0.0f;
    red[tid] = d * d; __syncthreads();
    for (int s = 128; s > 0; s >>= 1) { if (tid < s) red[tid] += red[tid + s]; __syncthreads(); }
    if (tid == 0) s_std = sqrtf(red[0] / (float)SEQ_ + 1e-5f);
    __syncthreads();
    float sd = s_std;
    if (tid < SEQ_) {
        float prev = (tid == 0) ? x_enc[b * SEQ_] : x_enc[b * SEQ_ + tid - 1];
        teacher[b * SEQ_ + tid] = (prev - m) / sd;
    }
    if (tid == 0) { nmean[b] = m; nstd[b] = sd; }
}

// ---------------------------------------------------------------------------
// Kernel 2: pack weights to bf16 [W_ih | zero pad | W_hh] row-major, fuse bias
// ---------------------------------------------------------------------------
__global__ void k_pack(const float* __restrict__ Wih0, const float* __restrict__ Whh0,
                       const float* __restrict__ bih0, const float* __restrict__ bhh0,
                       const float* __restrict__ Wih1, const float* __restrict__ Whh1,
                       const float* __restrict__ bih1, const float* __restrict__ bhh1,
                       const float* __restrict__ meanW, const float* __restrict__ meanB,
                       unsigned short* __restrict__ W0p, unsigned short* __restrict__ W1p,
                       float* __restrict__ b0, float* __restrict__ b1,
                       float* __restrict__ mWb) {
    const int n = blockIdx.x, tid = threadIdx.x;
    for (int k = tid; k < K0_; k += 256) {
        float w = (k < 67) ? Wih0[n * 67 + k] : ((k < 96) ? 0.0f : Whh0[n * HID_ + (k - 96)]);
        W0p[n * K0_ + k] = f2bf(w);
    }
    for (int k = tid; k < K1_; k += 256) {
        float w = (k < HID_) ? Wih1[n * HID_ + k] : Whh1[n * HID_ + (k - HID_)];
        W1p[n * K1_ + k] = f2bf(w);
    }
    if (tid == 0) { b0[n] = bih0[n] + bhh0[n]; b1[n] = bih1[n] + bhh1[n]; }
    if (n == 0) {
        if (tid < HID_) mWb[tid] = meanW[tid];
        if (tid == HID_) mWb[HID_] = meanB[0];
    }
}

// ---------------------------------------------------------------------------
// Kernel 3: persistent 2-layer LSTM. 64 WGs x 16 batch rows, 8 waves x 64 gate
// cols. bf16 WMMA 16x16x32, A in LDS, B streamed from L2-resident packed wts.
// ---------------------------------------------------------------------------
__global__ void __launch_bounds__(256, 1)
k_lstm(const float* __restrict__ given_enc, const float* __restrict__ x_mark,
       const float* __restrict__ meta_x,
       const float* __restrict__ te0, const float* __restrict__ te1, const float* __restrict__ te2,
       const float* __restrict__ me0, const float* __restrict__ me1,
       const float* __restrict__ teacher, const float* __restrict__ nmean,
       const float* __restrict__ nstd,
       const unsigned short* __restrict__ W0p, const unsigned short* __restrict__ W1p,
       const float* __restrict__ b0g, const float* __restrict__ b1g,
       const float* __restrict__ mWbg,
       float* __restrict__ out) {
    __shared__ unsigned short Xb[16 * K0_];    // layer-0 A: [feat(0..95) | h1(96..223)]
    __shared__ unsigned short X1b[16 * K1_];   // layer-1 A: [h1 | h2]
    __shared__ float Gb[G_ * 16];              // gates, [n][m] layout
    __shared__ float pmean[16];
    __shared__ float mAcc[16];
    __shared__ float b0s[G_], b1s[G_];
    __shared__ float mWs[HID_ + 4];

    const int tid   = threadIdx.x;
    const int wave  = tid >> 5;
    const int lane  = tid & 31;
    const int lh    = lane >> 4;     // K-half selector
    const int ln    = lane & 15;     // A: M row / B: N col / C: N col
    const int bbase = blockIdx.x * 16;

    // ---- one-time init ----
    for (int i = tid; i < 16 * K0_; i += 256) { if ((i % K0_) >= 67) Xb[i] = 0; }
    for (int i = tid; i < 16 * K1_; i += 256) X1b[i] = 0;
    for (int i = tid; i < G_; i += 256) { b0s[i] = b0g[i]; b1s[i] = b1g[i]; }
    for (int i = tid; i <= HID_; i += 256) mWs[i] = mWbg[i];
    if (tid < 16) { pmean[tid] = 0.0f; mAcc[tid] = 0.0f; }
    // static meta features (slots 33..66), constant over time
    for (int i = tid; i < 16 * 34; i += 256) {
        int m = i / 34, s = 33 + (i % 34);
        int b = bbase + m;
        float v;
        if (s < 35)      v = meta_x[(size_t)b * 4 + (s - 33)];
        else if (s < 51) v = me0[(size_t)((int)meta_x[(size_t)b * 4 + 2]) * 16 + (s - 35)];
        else             v = me1[(size_t)((int)meta_x[(size_t)b * 4 + 3]) * 16 + (s - 51)];
        Xb[m * K0_ + s] = f2bf(v);
    }
    float myStd = 1.0f, myMean = 0.0f;
    if (tid < 16) { myStd = nstd[bbase + tid]; myMean = nmean[bbase + tid]; }

    // elementwise-update ownership: thread -> (row m, 8 hidden cols), c in regs
    const int em    = tid & 15;
    const int jbase = (tid >> 4) * 8;
    float c1r[8], c2r[8];
#pragma unroll
    for (int e = 0; e < 8; ++e) { c1r[e] = 0.0f; c2r[e] = 0.0f; }

    __syncthreads();

#pragma unroll 1
    for (int t = 0; t < W_; ++t) {
        // ---- phase A: time-varying features, slots 0..32 ----
        for (int i = tid; i < 16 * 33; i += 256) {
            int m = i / 33, s = i % 33;
            int b = bbase + m;
            float v;
            if (s == 0) {
                v = (t < SEQ_) ? teacher[(size_t)b * SEQ_ + t] : pmean[m];
            } else if (s < 5) {
                v = given_enc[(size_t)b * (W_ * 7) + (size_t)t * 7 + (s - 1)];
            } else if (s < 29) {
                int tab = (s - 5) >> 3, pos = (s - 5) & 7;
                int idx = (int)given_enc[(size_t)b * (W_ * 7) + (size_t)t * 7 + 4 + tab];
                const float* tbl = (tab == 0) ? te0 : ((tab == 1) ? te1 : te2);
                v = tbl[(size_t)idx * 8 + pos];
            } else {
                v = x_mark[(size_t)b * (W_ * 4) + (size_t)t * 4 + (s - 29)];
            }
            Xb[m * K0_ + s] = f2bf(v);
        }
        if (tid < 16) mAcc[tid] = 0.0f;
        __syncthreads();

        // ---- phase B: layer-0 GEMM (16 x K0_) x (K0_ x 512) ----
        {
            const int n0 = wave * 64;
            v8f acc[4];
#pragma unroll
            for (int nt = 0; nt < 4; ++nt) {
                float bb = b0s[n0 + nt * 16 + ln];
#pragma unroll
                for (int v = 0; v < 8; ++v) acc[nt][v] = bb;
            }
#pragma unroll 1
            for (int kt = 0; kt < 7; ++kt) {
                const int kb = kt * 32;
                FragAB A;
                A.q[0] = *(const U128*)&Xb[ln * K0_ + kb + lh * 8];
                A.q[1] = *(const U128*)&Xb[ln * K0_ + kb + 16 + lh * 8];
#pragma unroll
                for (int nt = 0; nt < 4; ++nt) {
                    const int n = n0 + nt * 16 + ln;
                    const U128* wp = (const U128*)(W0p + (size_t)n * K0_ + kb + lh * 8);
                    FragAB Bv; Bv.q[0] = wp[0]; Bv.q[1] = wp[2];
                    acc[nt] = __builtin_amdgcn_wmma_f32_16x16x32_bf16(
                        false, A.v, false, Bv.v, (short)0, acc[nt], false, false);
                }
            }
#pragma unroll
            for (int nt = 0; nt < 4; ++nt) {
                const int n = n0 + nt * 16 + ln;
                FragC u; u.v = acc[nt];
                float* g = &Gb[n * 16 + lh * 8];
                *(F128*)g       = F128{u.f[0], u.f[1], u.f[2], u.f[3]};
                *(F128*)(g + 4) = F128{u.f[4], u.f[5], u.f[6], u.f[7]};
            }
        }
        __syncthreads();

        // ---- phase C: cell/hidden update layer 0 (hw tanh path) ----
#pragma unroll
        for (int e = 0; e < 8; ++e) {
            const int j = jbase + e;
            float gi = Gb[(j)        * 16 + em];
            float gf = Gb[(HID_ + j) * 16 + em];
            float gg = Gb[(2*HID_+j) * 16 + em];
            float go = Gb[(3*HID_+j) * 16 + em];
            c1r[e] = sigm(gf) * c1r[e] + sigm(gi) * ftanh(gg);
            float h = sigm(go) * ftanh(c1r[e]);
            unsigned short hb = f2bf(h);
            Xb [em * K0_ + 96 + j] = hb;   // recurrent input, layer 0
            X1b[em * K1_ + j]      = hb;   // input, layer 1
        }
        __syncthreads();

        // ---- phase D: layer-1 GEMM (16 x 256) x (256 x 512) ----
        {
            const int n0 = wave * 64;
            v8f acc[4];
#pragma unroll
            for (int nt = 0; nt < 4; ++nt) {
                float bb = b1s[n0 + nt * 16 + ln];
#pragma unroll
                for (int v = 0; v < 8; ++v) acc[nt][v] = bb;
            }
#pragma unroll 1
            for (int kt = 0; kt < 8; ++kt) {
                const int kb = kt * 32;
                FragAB A;
                A.q[0] = *(const U128*)&X1b[ln * K1_ + kb + lh * 8];
                A.q[1] = *(const U128*)&X1b[ln * K1_ + kb + 16 + lh * 8];
#pragma unroll
                for (int nt = 0; nt < 4; ++nt) {
                    const int n = n0 + nt * 16 + ln;
                    const U128* wp = (const U128*)(W1p + (size_t)n * K1_ + kb + lh * 8);
                    FragAB Bv; Bv.q[0] = wp[0]; Bv.q[1] = wp[2];
                    acc[nt] = __builtin_amdgcn_wmma_f32_16x16x32_bf16(
                        false, A.v, false, Bv.v, (short)0, acc[nt], false, false);
                }
            }
#pragma unroll
            for (int nt = 0; nt < 4; ++nt) {
                const int n = n0 + nt * 16 + ln;
                FragC u; u.v = acc[nt];
                float* g = &Gb[n * 16 + lh * 8];
                *(F128*)g       = F128{u.f[0], u.f[1], u.f[2], u.f[3]};
                *(F128*)(g + 4) = F128{u.f[4], u.f[5], u.f[6], u.f[7]};
            }
        }
        __syncthreads();

        // ---- phase E: cell/hidden update layer 1 + mean-head partial dot ----
        {
            float part = 0.0f;
#pragma unroll
            for (int e = 0; e < 8; ++e) {
                const int j = jbase + e;
                float gi = Gb[(j)        * 16 + em];
                float gf = Gb[(HID_ + j) * 16 + em];
                float gg = Gb[(2*HID_+j) * 16 + em];
                float go = Gb[(3*HID_+j) * 16 + em];
                c2r[e] = sigm(gf) * c2r[e] + sigm(gi) * ftanh(gg);
                float h = sigm(go) * ftanh(c2r[e]);
                X1b[em * K1_ + HID_ + j] = f2bf(h);   // recurrent input, layer 1
                part += h * mWs[j];
            }
            atomicAdd(&mAcc[em], part);               // ds_add_f32
        }
        __syncthreads();

        // ---- phase F: finalize mean, feedback + output ----
        if (tid < 16) {
            float mv = mAcc[tid] + mWs[HID_];
            pmean[tid] = mv;                          // normalized-space feedback
            if (t >= SEQ_)
                out[(size_t)(bbase + tid) * PRED_ + (t - SEQ_)] = mv * myStd + myMean;
        }
        __syncthreads();
    }
}

// ---------------------------------------------------------------------------
extern "C" void kernel_launch(void* const* d_in, const int* in_sizes, int n_in,
                              void* d_out, int out_size, void* d_ws, size_t ws_size,
                              hipStream_t stream) {
    const float* given_enc = (const float*)d_in[0];
    const float* x_enc     = (const float*)d_in[1];
    const float* x_mark    = (const float*)d_in[2];
    const float* meta_x    = (const float*)d_in[3];
    const float* te0 = (const float*)d_in[4];
    const float* te1 = (const float*)d_in[5];
    const float* te2 = (const float*)d_in[6];
    const float* me0 = (const float*)d_in[7];
    const float* me1 = (const float*)d_in[8];
    const float* Wih0 = (const float*)d_in[9];
    const float* Whh0 = (const float*)d_in[10];
    const float* bih0 = (const float*)d_in[11];
    const float* bhh0 = (const float*)d_in[12];
    const float* Wih1 = (const float*)d_in[13];
    const float* Whh1 = (const float*)d_in[14];
    const float* bih1 = (const float*)d_in[15];
    const float* bhh1 = (const float*)d_in[16];
    const float* meanW = (const float*)d_in[17];
    const float* meanB = (const float*)d_in[18];
    float* out = (float*)d_out;

    // workspace carve-up (256-byte aligned slabs)
    char* ws = (char*)d_ws;
    size_t off = 0;
    auto alloc = [&](size_t bytes) { char* p = ws + off; off = (off + bytes + 255) & ~(size_t)255; return p; };
    float*          teacher = (float*)alloc((size_t)B_ * SEQ_ * 4);
    float*          nmean   = (float*)alloc((size_t)B_ * 4);
    float*          nstd    = (float*)alloc((size_t)B_ * 4);
    unsigned short* W0p     = (unsigned short*)alloc((size_t)G_ * K0_ * 2);
    unsigned short* W1p     = (unsigned short*)alloc((size_t)G_ * K1_ * 2);
    float*          b0      = (float*)alloc((size_t)G_ * 4);
    float*          b1      = (float*)alloc((size_t)G_ * 4);
    float*          mWb     = (float*)alloc((size_t)(HID_ + 4) * 4);

    k_norm<<<B_, 256, 0, stream>>>(x_enc, teacher, nmean, nstd);
    k_pack<<<G_, 256, 0, stream>>>(Wih0, Whh0, bih0, bhh0, Wih1, Whh1, bih1, bhh1,
                                   meanW, meanB, W0p, W1p, b0, b1, mWb);
    k_lstm<<<B_ / 16, 256, 0, stream>>>(given_enc, x_mark, meta_x,
                                        te0, te1, te2, me0, me1,
                                        teacher, nmean, nstd,
                                        W0p, W1p, b0, b1, mWb, out);
}
